// GatedDeltaNetAttention_1082331758936
// MI455X (gfx1250) — matrix operand order repaired and checked by reading, not compiled, not run on hardware
//
#include <hip/hip_runtime.h>
#include <cstdint>
#include <cstddef>

typedef __attribute__((ext_vector_type(16))) __bf16 v16bf;
typedef __attribute__((ext_vector_type(8)))  float  v8f;
typedef __attribute__((ext_vector_type(4)))  unsigned int v4u;

#define CDIV(a,b) (((a)+(b)-1)/(b))

static const int LSEQ = 1024;
static const int D_IN = 2048;
static const int NH   = 16;
static const int HD   = 128;
static const int CKEY = 2048;
static const int CVAL = 2048;
static const int CCONV= 6144;
static const int GKT  = 64;     // K-tiles: all GEMMs here have K = 2048 = 64*32

__device__ __forceinline__ float sigmoidf_(float x){ return 1.f/(1.f+__expf(-x)); }
__device__ __forceinline__ float siluf_(float x){ return x * sigmoidf_(x); }

// ---------------------------------------------------------------------------
// Pack A: f32 row-major [M x K] -> bf16 WMMA A-tiles (16x32), tile t = mt*KT+kt
// A(m,k): lane = m + 16*((k>>3)&1), elem j: k = kbase + j + (j>=8 ? 8 : 0)
// ---------------------------------------------------------------------------
__global__ void pack_a_kernel(const float* __restrict__ src, __bf16* __restrict__ dst,
                              int KT, int ld, int total)
{
  int tid = blockIdx.x * blockDim.x + threadIdx.x;
  if (tid >= total) return;
  int e = tid & 511, t = tid >> 9;
  int lane = e >> 4, j = e & 15;
  int mt = t / KT, kt = t - mt * KT;
  int m = lane & 15;
  int kbase = (lane >> 4) << 3;
  int k = kbase + j + ((j >= 8) ? 8 : 0);
  dst[tid] = (__bf16)src[(size_t)(mt * 16 + m) * (size_t)ld + (size_t)(kt * 32 + k)];
}

// ---------------------------------------------------------------------------
// Pack B: f32 row-major [K x N] -> bf16 WMMA B-tiles (32x16), tile t = nt*KT+kt
// B(k,n): lane = n + 16*((k>>3)&1), elem j: k = kbase + j + (j>=8 ? 8 : 0)
// ---------------------------------------------------------------------------
__global__ void pack_b_kernel(const float* __restrict__ src, __bf16* __restrict__ dst,
                              int KT, int N, int total)
{
  int tid = blockIdx.x * blockDim.x + threadIdx.x;
  if (tid >= total) return;
  int e = tid & 511, t = tid >> 9;
  int lane = e >> 4, j = e & 15;
  int nt = t / KT, kt = t - nt * KT;
  int n = lane & 15;
  int k = ((lane >> 4) << 3) + j + ((j >= 8) ? 8 : 0);
  dst[tid] = (__bf16)src[(size_t)(kt * 32 + k) * (size_t)N + (size_t)(nt * 16 + n)];
}

// ---------------------------------------------------------------------------
// GEMM helpers (K = 2048 hardcoded -> compile-time LDS offsets)
// ---------------------------------------------------------------------------
__device__ __forceinline__ void gemm_prefetch(const char* gA, const char* gB, int kt,
                                              const int* cTile, const int* cOff, v4u* pre)
{
  #pragma unroll
  for (int r = 0; r < 3; ++r) {
    int t = cTile[r];
    const char* src = (t < 4)
      ? gA + ((size_t)t * GKT + kt) * 1024 + cOff[r]
      : gB + ((size_t)(t - 4) * GKT + kt) * 1024 + cOff[r];
    pre[r] = *(const v4u*)src;
  }
}

__device__ __forceinline__ void gemm_stage(char* sAbuf, char* sBbuf,
                                           const int* cTile, const int* cOff, const v4u* pre)
{
  #pragma unroll
  for (int r = 0; r < 3; ++r) {
    int t = cTile[r];
    char* dst = (t < 4) ? sAbuf + t * 1024 + cOff[r]
                        : sBbuf + (t - 4) * 1024 + cOff[r];
    *(v4u*)dst = pre[r];
  }
}

__device__ __forceinline__ void gemm_step(const char* sAbuf, const char* sBbuf,
                                          int wm, int wn, int lane,
                                          v8f& acc0, v8f& acc1, v8f& acc2, v8f& acc3)
{
  const char* pA = sAbuf + wm * 1024 + lane * 32;
  const char* pB = sBbuf + (wn * 4) * 1024 + lane * 32;
  v16bf a  = *(const v16bf*)pA;
  v16bf b0 = *(const v16bf*)(pB + 0 * 1024);
  v16bf b1 = *(const v16bf*)(pB + 1 * 1024);
  v16bf b2 = *(const v16bf*)(pB + 2 * 1024);
  v16bf b3 = *(const v16bf*)(pB + 3 * 1024);
  acc0 = __builtin_amdgcn_wmma_f32_16x16x32_bf16(false, a, false, b0, (short)0, acc0, false, false);
  acc1 = __builtin_amdgcn_wmma_f32_16x16x32_bf16(false, a, false, b1, (short)0, acc1, false, false);
  acc2 = __builtin_amdgcn_wmma_f32_16x16x32_bf16(false, a, false, b2, (short)0, acc2, false, false);
  acc3 = __builtin_amdgcn_wmma_f32_16x16x32_bf16(false, a, false, b3, (short)0, acc3, false, false);
}

// ---------------------------------------------------------------------------
// LDS-staged GEMM: C[MxN] = A * B, bf16 packed tiles, f32 accum via WMMA.
// Block = 256 threads = 8 waves in a 4(M) x 2(N) grid -> 64 x 128 output block.
// Double-buffered LDS staging, unrolled x2 so buffer indices are compile-time.
// Per k-step: 4 A-tiles + 8 B-tiles staged (12 KB) for 32 block-WMMAs.
// ---------------------------------------------------------------------------
__global__ void __launch_bounds__(256)
gemm_bf16_kernel(const __bf16* __restrict__ A, const __bf16* __restrict__ B,
                 float* __restrict__ C, int M, int N)
{
  const int NB = N >> 7;                 // 128-col blocks
  int mb = blockIdx.x / NB;
  int nb = blockIdx.x - mb * NB;
  if (mb >= (M >> 6)) return;

  int tid  = threadIdx.x;
  int lane = tid & 31;
  int w    = tid >> 5;                   // 0..7
  int wm   = w >> 1;                     // 0..3 (M)
  int wn   = w & 1;                      // 0..1 (N)

  __shared__ __bf16 sA[2][4 * 512];      // 4 KB per buffer
  __shared__ __bf16 sB[2][8 * 512];      // 8 KB per buffer

  const char* gA = (const char*)(A + (size_t)(mb * 4) * GKT * 512);
  const char* gB = (const char*)(B + (size_t)(nb * 8) * GKT * 512);

  // copy plan: 12 tiles * 1 KB = 768 16B-chunks; 3 chunks per thread
  int cTile[3], cOff[3];
  #pragma unroll
  for (int r = 0; r < 3; ++r) {
    int c = tid + 256 * r;
    cTile[r] = c >> 6;                   // 0..11  (0..3 = A, 4..11 = B)
    cOff[r]  = (c & 63) * 16;            // byte offset within tile
  }

  v8f acc0 = {0.f,0.f,0.f,0.f,0.f,0.f,0.f,0.f};
  v8f acc1 = acc0, acc2 = acc0, acc3 = acc0;

  v4u pre[3];

  // prologue: stage kt = 0 into buffer 0
  gemm_prefetch(gA, gB, 0, cTile, cOff, pre);
  gemm_stage((char*)&sA[0][0], (char*)&sB[0][0], cTile, cOff, pre);
  __syncthreads();

  for (int kt = 0; kt < GKT; kt += 2) {
    // even step: compute buf0, stage kt+1 -> buf1
    gemm_prefetch(gA, gB, kt + 1, cTile, cOff, pre);          // kt+1 <= 63 always
    gemm_step((const char*)&sA[0][0], (const char*)&sB[0][0], wm, wn, lane,
              acc0, acc1, acc2, acc3);
    __syncthreads();
    gemm_stage((char*)&sA[1][0], (char*)&sB[1][0], cTile, cOff, pre);
    __syncthreads();

    // odd step: compute buf1, stage kt+2 -> buf0
    bool more = (kt + 2 < GKT);
    if (more) gemm_prefetch(gA, gB, kt + 2, cTile, cOff, pre);
    gemm_step((const char*)&sA[1][0], (const char*)&sB[1][0], wm, wn, lane,
              acc0, acc1, acc2, acc3);
    __syncthreads();
    if (more) gemm_stage((char*)&sA[0][0], (char*)&sB[0][0], cTile, cOff, pre);
    __syncthreads();
  }

  // C/D layout: lanes 0-15 -> n = lane, rows r; lanes 16-31 -> rows r+8
  int m0 = mb * 64 + wm * 16 + ((lane >> 4) << 3);
  int nl = lane & 15;
  float* Cp = C + (size_t)m0 * N + (size_t)(nb * 128 + wn * 64 + nl);
  #pragma unroll
  for (int r = 0; r < 8; ++r) {
    Cp[(size_t)r * N +  0] = acc0[r];
    Cp[(size_t)r * N + 16] = acc1[r];
    Cp[(size_t)r * N + 32] = acc2[r];
    Cp[(size_t)r * N + 48] = acc3[r];
  }
}

// ---------------------------------------------------------------------------
// Causal depthwise conv (K=4) over L + SiLU.  y[l,c] = silu(sum_t w[c,t]*x[l-3+t,c])
// ---------------------------------------------------------------------------
__global__ void conv_silu_kernel(const float* __restrict__ qkv, const float* __restrict__ w,
                                 float* __restrict__ outc)
{
  int idx = blockIdx.x * blockDim.x + threadIdx.x;
  if (idx >= LSEQ * CCONV) return;
  int l = idx / CCONV, c = idx - l * CCONV;
  float s = 0.f;
  #pragma unroll
  for (int t = 0; t < 4; ++t) {
    int ls = l - 3 + t;
    if (ls >= 0) s += w[c * 4 + t] * qkv[(size_t)ls * CCONV + c];
  }
  outc[idx] = siluf_(s);
}

// ---------------------------------------------------------------------------
// In-place l2norm of q and k heads; q additionally scaled by DK^-0.5.
// One wave per (l, h, which).
// ---------------------------------------------------------------------------
__global__ void l2norm_kernel(float* __restrict__ qkvc)
{
  int gw = (blockIdx.x * blockDim.x + threadIdx.x) >> 5;
  int lane = threadIdx.x & 31;
  int which = gw & 1;
  int lh = gw >> 1;
  int h = lh & 15, l = lh >> 4;
  if (l >= LSEQ) return;
  float* p = qkvc + (size_t)l * CCONV + (size_t)which * CKEY + (size_t)h * HD;
  float v0 = p[lane], v1 = p[lane+32], v2 = p[lane+64], v3 = p[lane+96];
  float ss = v0*v0 + v1*v1 + v2*v2 + v3*v3;
  #pragma unroll
  for (int m = 16; m >= 1; m >>= 1) ss += __shfl_xor(ss, m, 32);
  float sc = rsqrtf(ss + 1e-6f);
  if (which == 0) sc *= 0.08838834764831845f;  // 1/sqrt(128)
  p[lane] = v0*sc; p[lane+32] = v1*sc; p[lane+64] = v2*sc; p[lane+96] = v3*sc;
}

// ---------------------------------------------------------------------------
// beta = sigmoid(x@Wb), dec = exp(-exp(A_log) * softplus(x@Wa + dt_bias)).
// Decay pre-exponentiated here to keep transcendentals out of the scan loop.
// One wave per (l, h), dot over IDIM=2048.
// ---------------------------------------------------------------------------
__global__ void betag_kernel(const float* __restrict__ x, const float* __restrict__ Wb,
                             const float* __restrict__ Wa, const float* __restrict__ A_log,
                             const float* __restrict__ dt_bias,
                             float* __restrict__ beta, float* __restrict__ dec)
{
  int gw = (blockIdx.x * blockDim.x + threadIdx.x) >> 5;
  int lane = threadIdx.x & 31;
  int h = gw & 15, l = gw >> 4;
  if (l >= LSEQ) return;
  const float* xr = x + (size_t)l * D_IN;
  float db = 0.f, da = 0.f;
  for (int d = lane; d < D_IN; d += 32) {
    float xv = xr[d];
    db += xv * Wb[(size_t)d * NH + h];
    da += xv * Wa[(size_t)d * NH + h];
  }
  #pragma unroll
  for (int m = 16; m >= 1; m >>= 1) { db += __shfl_xor(db, m, 32); da += __shfl_xor(da, m, 32); }
  if (lane == 0) {
    float dtv = da + dt_bias[h];
    float sp = (dtv > 20.f) ? dtv : log1pf(__expf(dtv));
    beta[l * NH + h] = sigmoidf_(db);
    dec[l * NH + h]  = __expf(-__expf(A_log[h]) * sp);
  }
}

// ---------------------------------------------------------------------------
// Sequential delta-rule scan. One block per head; 512 threads.
// Thread layout: wave wv (0..15), lane (0..31); column v = wv*8 + (lane>>2),
// d-group dg = lane&3 owns state rows d = dg*32 .. dg*32+31 for column v.
// The 4 partial owners of a column sit in adjacent lanes -> cross-group
// reductions are two shfl_xor ops (no barrier, no LDS round-trip).
// 2 barriers per step (k/q tile staging only).
// ---------------------------------------------------------------------------
__global__ void __launch_bounds__(512)
scan_kernel(const float* __restrict__ qkvc, const float* __restrict__ beta,
            const float* __restrict__ dec, float* __restrict__ attn)
{
  const int h    = blockIdx.x;
  const int t    = threadIdx.x;
  const int lane = t & 31;
  const int wv   = t >> 5;
  const int v    = wv * 8 + (lane >> 2);
  const int dg   = lane & 3;

  __shared__ float kq[256];   // k[0:128], q[128:256]

  float M[32];
  #pragma unroll
  for (int i = 0; i < 32; ++i) M[i] = 0.f;

  const float* qb = qkvc + (size_t)h * HD;
  const float* kb = qkvc + CKEY + (size_t)h * HD;
  const float* vb = qkvc + 2 * (size_t)CKEY + (size_t)h * HD;

  for (int l = 0; l < LSEQ; ++l) {
    size_t rowoff = (size_t)l * CCONV;
    if (t < 128)       kq[t] = kb[rowoff + t];
    else if (t < 256)  kq[t] = qb[rowoff + (t - 128)];
    __syncthreads();

    float dc = dec[l * NH + h];
    float bt = beta[l * NH + h];
    float vt = vb[rowoff + v];
    const float* kl = &kq[dg * 32];
    const float* ql = &kq[128 + dg * 32];

    float pk = 0.f;
    #pragma unroll
    for (int i = 0; i < 32; ++i) { M[i] *= dc; pk += kl[i] * M[i]; }
    pk += __shfl_xor(pk, 1, 32);
    pk += __shfl_xor(pk, 2, 32);        // full Mk in all 4 lanes of the group
    float err = (vt - pk) * bt;

    float pq = 0.f;
    #pragma unroll
    for (int i = 0; i < 32; ++i) { M[i] += kl[i] * err; pq += ql[i] * M[i]; }
    pq += __shfl_xor(pq, 1, 32);
    pq += __shfl_xor(pq, 2, 32);
    if (dg == 0)
      attn[(size_t)l * CVAL + (size_t)h * HD + v] = pq;
    __syncthreads();                    // protect kq before next step's staging
  }
}

// ---------------------------------------------------------------------------
// Gated RMSNorm: gated = out * rsqrt(mean(out^2)+eps) * norm_w * silu(z)
// ---------------------------------------------------------------------------
__global__ void gate_kernel(const float* __restrict__ attn, const float* __restrict__ z,
                            const float* __restrict__ norm_w, float* __restrict__ gated)
{
  int gw = (blockIdx.x * blockDim.x + threadIdx.x) >> 5;
  int lane = threadIdx.x & 31;
  int h = gw & 15, l = gw >> 4;
  if (l >= LSEQ) return;
  const float* o  = attn + (size_t)l * CVAL + (size_t)h * HD;
  const float* zr = z    + (size_t)l * CVAL + (size_t)h * HD;
  float v0=o[lane], v1=o[lane+32], v2=o[lane+64], v3=o[lane+96];
  float ss = v0*v0 + v1*v1 + v2*v2 + v3*v3;
  #pragma unroll
  for (int m = 16; m >= 1; m >>= 1) ss += __shfl_xor(ss, m, 32);
  float rms = rsqrtf(ss * (1.f / 128.f) + 1e-6f);
  float* gp = gated + (size_t)l * CVAL + (size_t)h * HD;
  gp[lane]    = v0 * rms * norm_w[lane]    * siluf_(zr[lane]);
  gp[lane+32] = v1 * rms * norm_w[lane+32] * siluf_(zr[lane+32]);
  gp[lane+64] = v2 * rms * norm_w[lane+64] * siluf_(zr[lane+64]);
  gp[lane+96] = v3 * rms * norm_w[lane+96] * siluf_(zr[lane+96]);
}

// ---------------------------------------------------------------------------
extern "C" void kernel_launch(void* const* d_in, const int* in_sizes, int n_in,
                              void* d_out, int out_size, void* d_ws, size_t ws_size,
                              hipStream_t stream)
{
  (void)in_sizes; (void)n_in; (void)out_size; (void)ws_size;
  const float* x       = (const float*)d_in[0];
  const float* Wqkv    = (const float*)d_in[1];
  const float* Wz      = (const float*)d_in[2];
  const float* Wb      = (const float*)d_in[3];
  const float* Wa      = (const float*)d_in[4];
  const float* conv_w  = (const float*)d_in[5];
  const float* A_log   = (const float*)d_in[6];
  const float* dt_bias = (const float*)d_in[7];
  const float* norm_w  = (const float*)d_in[8];
  const float* Wout    = (const float*)d_in[9];
  float* out = (float*)d_out;
  char*  ws  = (char*)d_ws;

  // workspace layout (bytes)
  const size_t o_xA     = 0;                      // 1024x2048 bf16 packed A   (4 MB)
  const size_t o_WqkvB  = 4194304;                // 2048x6144 bf16 packed B   (24 MB)
  const size_t o_WzB    = 29360128;               // 2048x2048 bf16 packed B   (8 MB)
  const size_t o_WoutB  = 37748736;               // 2048x2048 bf16 packed B   (8 MB)
  const size_t o_qkv    = 46137344;               // raw qkv f32 (24 MB) — recycled below
  const size_t o_qkvc   = 71303168;               // post-conv qkv f32 (24 MB)
  const size_t o_z      = 96468992;               // z f32 (8 MB)
  const size_t o_beta   = 104857600;              // 64 KB
  const size_t o_dec    = 104923136;              // 64 KB
  // recycled region (raw qkv dead after conv):
  const size_t o_attn   = o_qkv;                  // 8 MB
  const size_t o_gated  = o_qkv + 8388608;        // 8 MB
  const size_t o_gatedA = o_qkv + 16777216;       // 4 MB

  __bf16* xA     = (__bf16*)(ws + o_xA);
  __bf16* WqkvB  = (__bf16*)(ws + o_WqkvB);
  __bf16* WzB    = (__bf16*)(ws + o_WzB);
  __bf16* WoutB  = (__bf16*)(ws + o_WoutB);
  __bf16* gatedA = (__bf16*)(ws + o_gatedA);
  float*  qkvf   = (float*)(ws + o_qkv);
  float*  qkvc   = (float*)(ws + o_qkvc);
  float*  zf     = (float*)(ws + o_z);
  float*  betaf  = (float*)(ws + o_beta);
  float*  decf   = (float*)(ws + o_dec);
  float*  attnf  = (float*)(ws + o_attn);
  float*  gatedf = (float*)(ws + o_gated);

  const int TB = 256;

  // 1) pack inputs to WMMA layouts (f32 -> bf16)
  { int tot = LSEQ * D_IN;
    pack_a_kernel<<<CDIV(tot,TB),TB,0,stream>>>(x, xA, D_IN/32, D_IN, tot); }
  { int tot = D_IN * CCONV;
    pack_b_kernel<<<CDIV(tot,TB),TB,0,stream>>>(Wqkv, WqkvB, D_IN/32, CCONV, tot); }
  { int tot = D_IN * CVAL;
    pack_b_kernel<<<CDIV(tot,TB),TB,0,stream>>>(Wz, WzB, D_IN/32, CVAL, tot); }
  { int tot = CVAL * D_IN;
    pack_b_kernel<<<CDIV(tot,TB),TB,0,stream>>>(Wout, WoutB, CVAL/32, D_IN, tot); }

  // 2) qkv = x @ Wqkv  (1024 x 6144)
  { int blocks = (LSEQ/64) * (CCONV/128);
    gemm_bf16_kernel<<<blocks,TB,0,stream>>>(xA, WqkvB, qkvf, LSEQ, CCONV); }
  // 3) z = x @ Wz  (1024 x 2048)
  { int blocks = (LSEQ/64) * (CVAL/128);
    gemm_bf16_kernel<<<blocks,TB,0,stream>>>(xA, WzB, zf, LSEQ, CVAL); }

  // 4) causal depthwise conv + SiLU
  { int tot = LSEQ * CCONV;
    conv_silu_kernel<<<CDIV(tot,TB),TB,0,stream>>>(qkvf, conv_w, qkvc); }

  // 5) l2norm q,k (q also * DK^-0.5)
  { int waves = LSEQ * NH * 2;
    l2norm_kernel<<<CDIV(waves*32,TB),TB,0,stream>>>(qkvc); }

  // 6) beta, decay
  { int waves = LSEQ * NH;
    betag_kernel<<<CDIV(waves*32,TB),TB,0,stream>>>(x, Wb, Wa, A_log, dt_bias, betaf, decf); }

  // 7) sequential delta-rule scan
  scan_kernel<<<NH, 512, 0, stream>>>(qkvc, betaf, decf, attnf);

  // 8) gated RMSNorm
  { int waves = LSEQ * NH;
    gate_kernel<<<CDIV(waves*32,TB),TB,0,stream>>>(attnf, zf, norm_w, gatedf); }

  // 9) out = gated @ Wout
  { int tot = LSEQ * CVAL;
    pack_a_kernel<<<CDIV(tot,TB),TB,0,stream>>>(gatedf, gatedA, CVAL/32, CVAL, tot); }
  { int blocks = (LSEQ/64) * (D_IN/128);
    gemm_bf16_kernel<<<blocks,TB,0,stream>>>(gatedA, WoutB, out, LSEQ, D_IN); }
}
